// EMPSN_Rephine_Cont_30863634989085
// MI455X (gfx1250) — matrix operand order repaired and checked by reading, not compiled, not run on hardware
//
#include <hip/hip_runtime.h>

typedef __attribute__((ext_vector_type(16))) __bf16 v16bf;
typedef __attribute__((ext_vector_type(8)))  __bf16 v8bf;
typedef __attribute__((ext_vector_type(8)))  float  v8f;

__device__ __forceinline__ float sigmoidf_(float x) { return 1.0f / (1.0f + __expf(-x)); }

__device__ __forceinline__ v8f wmma_bf16(v16bf a, v16bf b, v8f c) {
    return __builtin_amdgcn_wmma_f32_16x16x32_bf16(false, a, false, b, (short)0, c, false, false);
}

// Load 16-bit A fragment (16x32 tile) from an LDS row-major bf16 tile.
// ISA layout: lane L (half = L>>4, row = L&15): elems 0..7 -> K = half*8 + e,
// elems 8..15 -> K = half*8 + 16 + (e-8). Two contiguous 16B LDS loads.
__device__ __forceinline__ v16bf load_a_frag(const __bf16* tile, int stride, int kt, int lane) {
    int row = lane & 15, half = lane >> 4;
    const __bf16* p = tile + row * stride + kt * 32 + half * 8;
    v8bf lo = *(const v8bf*)p;
    v8bf hi = *(const v8bf*)(p + 16);
    return __builtin_shufflevector(lo, hi, 0,1,2,3,4,5,6,7,8,9,10,11,12,13,14,15);
}

// B fragments are pre-packed so each lane loads 16 contiguous bf16 (32B).
__device__ __forceinline__ v16bf load_b_frag(const __bf16* packed, int fragIdx, int lane) {
    return ((const v16bf*)packed)[fragIdx * 32 + lane];
}

__global__ void copy_f32(float* __restrict__ dst, const float* __restrict__ src, int n) {
    int i = blockIdx.x * 256 + threadIdx.x;
    if (i < n) dst[i] = src[i];
}

__global__ void zero_f32(float* __restrict__ dst, int n) {
    int i = blockIdx.x * 256 + threadIdx.x;
    if (i < n) dst[i] = 0.0f;
}

// Pack a KxN fp32 weight into fragment-native bf16:
// dst[((kt*NT+nt)*32 + L)*16 + e] = W[kt*32 + (L>>4)*8 + (e&7) + (e>>3)*16][nt*16 + (L&15)], zero padded.
__global__ void pack_frag(const float* __restrict__ src, __bf16* __restrict__ dst,
                          int K, int N, int KT, int NT) {
    int o = blockIdx.x * 256 + threadIdx.x;
    if (o >= KT * NT * 512) return;
    int e = o & 15, L = (o >> 4) & 31, frag = o >> 9;
    int nt = frag % NT, kt = frag / NT;
    int k = kt * 32 + (L >> 4) * 8 + (e & 7) + (e >> 3) * 16;
    int n = nt * 16 + (L & 15);
    float v = (k < K && n < N) ? src[k * N + n] : 0.0f;
    dst[o] = (__bf16)v;
}

// Fused EGNN edge pass: gather -> [MLP 133/136->65->65 silu] -> sigmoid gate -> atomic scatter.
// One wave per 16-edge tile; 4 independent waves per block (private LDS slices).
__global__ __launch_bounds__(128) void edge_kernel(
    const float* __restrict__ xsend, const float* __restrict__ xrec,
    const int* __restrict__ isend, const int* __restrict__ irec,
    const float* __restrict__ inv, int inv_dim, int E,
    const __bf16* __restrict__ W1p, const __bf16* __restrict__ W2p,
    const float* __restrict__ b1, const float* __restrict__ b2,
    const float* __restrict__ We, const float* __restrict__ be,
    float* __restrict__ mes, float tval)
{
    __shared__ __align__(16) __bf16 sA[4][16 * 160];
    __shared__ __align__(16) __bf16 sM[4][16 * 96];
    __shared__ int sRec[4][16];
    __shared__ int sSend[4][16];

    int wave = threadIdx.x >> 5, lane = threadIdx.x & 31;
    int tile = blockIdx.x * 4 + wave;
    int e0 = tile * 16;
    if (e0 >= E) return;
    __bf16* A = sA[wave];
    __bf16* M = sM[wave];

    if (lane < 16) {
        int e = e0 + lane, s = -1, r = -1;
        if (e < E) { s = isend[e]; r = irec[e]; }
        sSend[wave][lane] = s;
        sRec[wave][lane] = r;
    }
    // Stage state = [x_send(64) | t | x_rec(64) | t | inv | 0-pad] as bf16, K padded to 160.
    for (int i = lane; i < 16 * 160; i += 32) {
        int row = i / 160, col = i - row * 160;
        float f = 0.0f;
        int e = e0 + row;
        if (e < E) {
            int s = sSend[wave][row], r = sRec[wave][row];
            if (col < 64)                    f = xsend[(long)s * 64 + col];
            else if (col == 64)              f = tval;
            else if (col < 129)              f = xrec[(long)r * 64 + (col - 65)];
            else if (col == 129)             f = tval;
            else if (col < 130 + inv_dim)    f = inv[(long)e * inv_dim + (col - 130)];
        }
        A[row * 160 + col] = (__bf16)f;
    }

    int half = lane >> 4, l15 = lane & 15;

    // Layer 1: D(16x80) = A(16x160) x W1(160x80)
    v8f acc[5];
#pragma unroll
    for (int nt = 0; nt < 5; nt++) acc[nt] = (v8f){0,0,0,0,0,0,0,0};
    for (int kt = 0; kt < 5; kt++) {
        v16bf a = load_a_frag(A, 160, kt, lane);
#pragma unroll
        for (int nt = 0; nt < 5; nt++)
            acc[nt] = wmma_bf16(a, load_b_frag(W1p, kt * 5 + nt, lane), acc[nt]);
    }
    // SiLU(+b1), retile C-layout -> A-layout (LDS, K padded to 96)
#pragma unroll
    for (int nt = 0; nt < 5; nt++) {
        int col = nt * 16 + l15;
        float bb = (col < 65) ? b1[col] : 0.0f;
#pragma unroll
        for (int r = 0; r < 8; r++) {
            float f = acc[nt][r] + bb;
            M[(r + half * 8) * 96 + col] = (__bf16)(f * sigmoidf_(f));
        }
    }
    for (int i = lane; i < 16 * 16; i += 32)
        M[(i >> 4) * 96 + 80 + (i & 15)] = (__bf16)0.0f;

    // Layer 2: D(16x80) = m(16x96) x W2(96x80)
    v8f acc2[5];
#pragma unroll
    for (int nt = 0; nt < 5; nt++) acc2[nt] = (v8f){0,0,0,0,0,0,0,0};
    for (int kt = 0; kt < 3; kt++) {
        v16bf a = load_a_frag(M, 96, kt, lane);
#pragma unroll
        for (int nt = 0; nt < 5; nt++)
            acc2[nt] = wmma_bf16(a, load_b_frag(W2p, kt * 5 + nt, lane), acc2[nt]);
    }
    float be0 = be[0];
#pragma unroll
    for (int nt = 0; nt < 5; nt++) {
        int col = nt * 16 + l15;
        float bb = (col < 65) ? b2[col] : 0.0f;
#pragma unroll
        for (int r = 0; r < 8; r++) {
            float f = acc2[nt][r] + bb;
            acc2[nt][r] = f * sigmoidf_(f);   // m2
        }
    }
    // Per-row gate = sigmoid(m2 . We + be); reduce over the 16 lanes of each half.
#pragma unroll
    for (int r = 0; r < 8; r++) {
        float p = 0.0f;
#pragma unroll
        for (int nt = 0; nt < 5; nt++) {
            int col = nt * 16 + l15;
            p += acc2[nt][r] * ((col < 65) ? We[col] : 0.0f);
        }
        p += __shfl_xor(p, 1, 32);
        p += __shfl_xor(p, 2, 32);
        p += __shfl_xor(p, 4, 32);
        p += __shfl_xor(p, 8, 32);
        float gate = sigmoidf_(p + be0);
        int Mrow = r + half * 8;
        int rec = sRec[wave][Mrow];
        if (rec >= 0) {
#pragma unroll
            for (int nt = 0; nt < 5; nt++) {
                int col = nt * 16 + l15;
                if (col < 65)
                    atomicAdd(&mes[(long)rec * 65 + col], acc2[nt][r] * gate);
            }
        }
    }
}

// Fused node update: h = silu([xt|mes..] U1 + ub1) U2 + ub2 ; dx = xt Ws + bs + h ; x += dt*dx
__global__ __launch_bounds__(128) void node_kernel(
    const float* __restrict__ x, int N, float tval,
    const float* __restrict__ mesA, const float* __restrict__ mesB, int nmes,
    const __bf16* __restrict__ U1p, int KT1,
    const __bf16* __restrict__ U2p, const __bf16* __restrict__ Wsp,
    const float* __restrict__ ub1, const float* __restrict__ ub2,
    const float* __restrict__ bs, float dt, float* __restrict__ dst)
{
    __shared__ __align__(16) __bf16 sA[4][16 * 224];
    __shared__ __align__(16) __bf16 sM[4][16 * 96];
    __shared__ __align__(16) __bf16 sX[4][16 * 96];

    int wave = threadIdx.x >> 5, lane = threadIdx.x & 31;
    int tile = blockIdx.x * 4 + wave;
    int n0 = tile * 16;
    if (n0 >= N) return;
    int S = KT1 * 32;                 // 160 (factor 2) or 224 (factor 3)
    __bf16* A = sA[wave];
    __bf16* M = sM[wave];
    __bf16* X = sX[wave];

    for (int i = lane; i < 16 * S; i += 32) {
        int row = i / S, col = i - row * S;
        int n = n0 + row;
        float f = 0.0f;
        if (n < N) {
            if (col < 64)                        f = x[(long)n * 64 + col];
            else if (col == 64)                  f = tval;
            else if (col < 130)                  f = mesA[(long)n * 65 + (col - 65)];
            else if (nmes == 2 && col < 195)     f = mesB[(long)n * 65 + (col - 130)];
        }
        A[row * S + col] = (__bf16)f;
    }
    for (int i = lane; i < 16 * 96; i += 32) {   // xt tile for the Ws skip path (K padded to 96)
        int row = i / 96, col = i - row * 96;
        int n = n0 + row;
        float f = 0.0f;
        if (n < N && col < 65) f = (col < 64) ? x[(long)n * 64 + col] : tval;
        X[row * 96 + col] = (__bf16)f;
    }

    int half = lane >> 4, l15 = lane & 15;

    v8f acc1[5];
#pragma unroll
    for (int nt = 0; nt < 5; nt++) acc1[nt] = (v8f){0,0,0,0,0,0,0,0};
    for (int kt = 0; kt < KT1; kt++) {
        v16bf a = load_a_frag(A, S, kt, lane);
#pragma unroll
        for (int nt = 0; nt < 5; nt++)
            acc1[nt] = wmma_bf16(a, load_b_frag(U1p, kt * 5 + nt, lane), acc1[nt]);
    }
#pragma unroll
    for (int nt = 0; nt < 5; nt++) {
        int col = nt * 16 + l15;
        float bb = (col < 65) ? ub1[col] : 0.0f;
#pragma unroll
        for (int r = 0; r < 8; r++) {
            float f = acc1[nt][r] + bb;
            M[(r + half * 8) * 96 + col] = (__bf16)(f * sigmoidf_(f));
        }
    }
    for (int i = lane; i < 16 * 16; i += 32)
        M[(i >> 4) * 96 + 80 + (i & 15)] = (__bf16)0.0f;

    v8f acc2[4];
#pragma unroll
    for (int nt = 0; nt < 4; nt++) acc2[nt] = (v8f){0,0,0,0,0,0,0,0};
    for (int kt = 0; kt < 3; kt++) {             // h @ U2
        v16bf a = load_a_frag(M, 96, kt, lane);
#pragma unroll
        for (int nt = 0; nt < 4; nt++)
            acc2[nt] = wmma_bf16(a, load_b_frag(U2p, kt * 4 + nt, lane), acc2[nt]);
    }
    for (int kt = 0; kt < 3; kt++) {             // + xt @ Ws
        v16bf a = load_a_frag(X, 96, kt, lane);
#pragma unroll
        for (int nt = 0; nt < 4; nt++)
            acc2[nt] = wmma_bf16(a, load_b_frag(Wsp, kt * 4 + nt, lane), acc2[nt]);
    }
#pragma unroll
    for (int nt = 0; nt < 4; nt++) {
        int col = nt * 16 + l15;
        float bb = ub2[col] + bs[col];
#pragma unroll
        for (int r = 0; r < 8; r++) {
            int n = n0 + r + half * 8;
            if (n < N) {
                float dx = acc2[nt][r] + bb;
                dst[(long)n * 64 + col] = x[(long)n * 64 + col] + dt * dx;
            }
        }
    }
}

#define GRID1(n) dim3(((n) + 255) / 256)

extern "C" void kernel_launch(void* const* d_in, const int* in_sizes, int n_in,
                              void* d_out, int out_size, void* d_ws, size_t ws_size,
                              hipStream_t stream) {
    const int N0 = 80000, N1 = 150000, N2 = 80000;
    const int E00 = 300000, E01 = 200000, E11 = 300000, E12 = 150000;

    const float* x0 = (const float*)d_in[0];
    const float* x1 = (const float*)d_in[1];
    const float* x2 = (const float*)d_in[2];
    float* out = (float*)d_out;

    char* ws = (char*)d_ws;
    float* xs0 = (float*)ws;
    float* xs1 = xs0 + (size_t)N0 * 64;
    float* xs2 = xs1 + (size_t)N1 * 64;
    size_t off = (size_t)(N0 + N1 + N2) * 64 * 4;
    float* mes00 = (float*)(ws + off); off += (size_t)N0 * 65 * 4;
    float* mes01 = (float*)(ws + off); off += (size_t)N1 * 65 * 4;
    float* mes11 = (float*)(ws + off); off += (size_t)N1 * 65 * 4;
    float* mes12 = (float*)(ws + off); off += (size_t)N2 * 65 * 4;
    off = (off + 63) & ~(size_t)63;
    __bf16* wb = (__bf16*)(ws + off);
    __bf16* w1p[4] = { wb, wb + 12800, wb + 25600, wb + 38400 };
    __bf16* w2p[4] = { wb + 51200, wb + 58880, wb + 66560, wb + 74240 };
    __bf16* u1p[3] = { wb + 81920, wb + 94720, wb + 112640 };
    __bf16* u2p[3] = { wb + 125440, wb + 131584, wb + 137728 };
    __bf16* wsp = wb + 143872;

    // ---- init: xs <- inputs ----
    copy_f32<<<GRID1(N0 * 64), 256, 0, stream>>>(xs0, x0, N0 * 64);
    copy_f32<<<GRID1(N1 * 64), 256, 0, stream>>>(xs1, x1, N1 * 64);
    copy_f32<<<GRID1(N2 * 64), 256, 0, stream>>>(xs2, x2, N2 * 64);

    // ---- pack weights (params flattened in jax tree order: sorted dict keys) ----
    // leaves: 15..20 (0_0: W1,W2,We,b1,b2,be), 21..26 (0_1), 27..32 (1_1), 33..38 (1_2),
    //         39 Ws, 40 bs, 41..44 (upd 0: U1,U2,ub1,ub2), 45..48 (upd 1), 49..52 (upd 2)
    const int w1K[4] = { 133, 133, 136, 136 };
    for (int a = 0; a < 4; a++) {
        pack_frag<<<GRID1(5 * 5 * 512), 256, 0, stream>>>((const float*)d_in[15 + 6 * a], w1p[a], w1K[a], 65, 5, 5);
        pack_frag<<<GRID1(3 * 5 * 512), 256, 0, stream>>>((const float*)d_in[16 + 6 * a], w2p[a], 65, 65, 3, 5);
    }
    const int u1K[3] = { 130, 195, 130 };
    const int u1KT[3] = { 5, 7, 5 };
    for (int d = 0; d < 3; d++) {
        pack_frag<<<GRID1(u1KT[d] * 5 * 512), 256, 0, stream>>>((const float*)d_in[41 + 4 * d], u1p[d], u1K[d], 65, u1KT[d], 5);
        pack_frag<<<GRID1(3 * 4 * 512), 256, 0, stream>>>((const float*)d_in[42 + 4 * d], u2p[d], 65, 64, 3, 4);
    }
    pack_frag<<<GRID1(3 * 4 * 512), 256, 0, stream>>>((const float*)d_in[39], wsp, 65, 64, 3, 4);

    const float dt = 0.5f;
    for (int s = 0; s < 2; s++) {
        float tv = 0.5f * (float)s;

        zero_f32<<<GRID1(N0 * 65), 256, 0, stream>>>(mes00, N0 * 65);
        zero_f32<<<GRID1(N1 * 65), 256, 0, stream>>>(mes01, N1 * 65);
        zero_f32<<<GRID1(N1 * 65), 256, 0, stream>>>(mes11, N1 * 65);
        zero_f32<<<GRID1(N2 * 65), 256, 0, stream>>>(mes12, N2 * 65);

        #define EGRID(E) dim3((((E) + 15) / 16 + 3) / 4)
        edge_kernel<<<EGRID(E00), 128, 0, stream>>>(xs0, xs0,
            (const int*)d_in[4], (const int*)d_in[5], (const float*)d_in[3], 3, E00,
            w1p[0], w2p[0], (const float*)d_in[18], (const float*)d_in[19],
            (const float*)d_in[17], (const float*)d_in[20], mes00, tv);
        edge_kernel<<<EGRID(E01), 128, 0, stream>>>(xs0, xs1,
            (const int*)d_in[7], (const int*)d_in[8], (const float*)d_in[6], 3, E01,
            w1p[1], w2p[1], (const float*)d_in[24], (const float*)d_in[25],
            (const float*)d_in[23], (const float*)d_in[26], mes01, tv);
        edge_kernel<<<EGRID(E11), 128, 0, stream>>>(xs1, xs1,
            (const int*)d_in[10], (const int*)d_in[11], (const float*)d_in[9], 6, E11,
            w1p[2], w2p[2], (const float*)d_in[30], (const float*)d_in[31],
            (const float*)d_in[29], (const float*)d_in[32], mes11, tv);
        edge_kernel<<<EGRID(E12), 128, 0, stream>>>(xs1, xs2,
            (const int*)d_in[13], (const int*)d_in[14], (const float*)d_in[12], 6, E12,
            w1p[3], w2p[3], (const float*)d_in[36], (const float*)d_in[37],
            (const float*)d_in[35], (const float*)d_in[38], mes12, tv);

        float* dst0 = (s == 0) ? xs0 : out;
        float* dst1 = (s == 0) ? xs1 : out + (size_t)N0 * 64;
        float* dst2 = (s == 0) ? xs2 : out + (size_t)(N0 + N1) * 64;
        #define NGRID(N) dim3((((N) + 15) / 16 + 3) / 4)
        node_kernel<<<NGRID(N0), 128, 0, stream>>>(xs0, N0, tv, mes00, nullptr, 1,
            u1p[0], 5, u2p[0], wsp, (const float*)d_in[43], (const float*)d_in[44],
            (const float*)d_in[40], dt, dst0);
        node_kernel<<<NGRID(N1), 128, 0, stream>>>(xs1, N1, tv, mes01, mes11, 2,
            u1p[1], 7, u2p[1], wsp, (const float*)d_in[47], (const float*)d_in[48],
            (const float*)d_in[40], dt, dst1);
        node_kernel<<<NGRID(N2), 128, 0, stream>>>(xs2, N2, tv, mes12, nullptr, 1,
            u1p[2], 5, u2p[2], wsp, (const float*)d_in[51], (const float*)d_in[52],
            (const float*)d_in[40], dt, dst2);
    }
}
